// DQN_2697239462702
// MI455X (gfx1250) — compile-verified, matrix operand order checked by reading, not run on hardware
//
#include <hip/hip_runtime.h>

typedef _Float16 v16h __attribute__((ext_vector_type(16)));
typedef _Float16 v8h  __attribute__((ext_vector_type(8)));
typedef _Float16 h2   __attribute__((ext_vector_type(2)));
typedef float    v8f  __attribute__((ext_vector_type(8)));
typedef float    f4   __attribute__((ext_vector_type(4)));
typedef float    f4u  __attribute__((ext_vector_type(4), aligned(4))); // x rows have odd (2049) stride

#define SEQ   2048
#define SX    2049
#define KT1   64    // 2048 / 32
#define NT1   8     // 128 / 16
#define KT2   4     // 128 / 32
#define KTF1  6     // 192 (176 padded) / 32
#define KTF2  4     // 128 / 32
#define FRAG  512   // halves per 32x16 (or 16x32) f16 fragment

// ---------------------------------------------------------------------------
// Repack an f32 row-major (K x N) weight into f16 WMMA B-fragments.
// Fragment (kt, nt) lives at dst[(nt*KT + kt)*FRAG]; per lane 16 halves:
// lane = 16*h + n16 :  column N = nt*16 + n16,  K = kt*32 + 8*h + {0..7, 16..23}
// ---------------------------------------------------------------------------
__global__ __launch_bounds__(32) void pack_kernel(const float* __restrict__ src,
                                                  _Float16* __restrict__ dst,
                                                  int Ktrue, int N, int KT) {
  int fid  = blockIdx.x;
  int kt   = fid % KT;
  int lane = threadIdx.x & 31;
  int n    = (fid / KT) * 16 + (lane & 15);
  int koff = (lane >> 4) * 8;
  _Float16* d = dst + (size_t)fid * FRAG + lane * 16;
#pragma unroll
  for (int i = 0; i < 16; ++i) {
    int k = kt * 32 + koff + (i < 8 ? i : i + 8);
    float v = (k < Ktrue) ? src[(size_t)k * N + n] : 0.f;
    d[i] = (_Float16)v;
  }
}

// ---------------------------------------------------------------------------
// Fused: per-row stats + layer1 GEMM (K=2048) + layer2 GEMM (K=128) + aux/up
// MLPs.  One wave = one 16-row M tile, all 8 N tiles resident (v8f c[8]).
// Single pass over x (537 MB) — the HBM-critical resource.
// ---------------------------------------------------------------------------
__global__ __launch_bounds__(256) void main_kernel(
    const float* __restrict__ x,
    const _Float16* __restrict__ W1h, const float* __restrict__ b1,
    const _Float16* __restrict__ W2h, const float* __restrict__ b2,
    const float* __restrict__ w_aux1, const float* __restrict__ b_aux1,
    const float* __restrict__ w_aux2, const float* __restrict__ b_aux2,
    const float* __restrict__ w_up1,  const float* __restrict__ b_up1,
    const float* __restrict__ w_up2,  const float* __restrict__ b_up2,
    _Float16* __restrict__ xcat)
{
  __shared__ float hs[8][16 * 128];     // 64 KB, wave-private slices
  const int wave = threadIdx.x >> 5;
  const int lane = threadIdx.x & 31;
  const int r    = lane & 15;
  const int kh   = lane >> 4;           // which K half-group this lane carries
  const int Rb   = (blockIdx.x * 8 + wave) * 16;
  const int row  = Rb + r;
  const float* xrow = x + (size_t)row * SX;

  v8f c[8] = {};
  float msum = 0.f, mcnt = 0.f, lastVal = 0.f;
  int lastIdx = -1;

  for (int kt = 0; kt < KT1; ++kt) {
    int k0 = kt * 32 + kh * 8;
    f4u x0 = *reinterpret_cast<const f4u*>(xrow + k0);
    f4u x1 = *reinterpret_cast<const f4u*>(xrow + k0 + 4);
    f4u x2 = *reinterpret_cast<const f4u*>(xrow + k0 + 16);
    f4u x3 = *reinterpret_cast<const f4u*>(xrow + k0 + 20);
    v16h a;
#pragma unroll
    for (int i = 0; i < 4; ++i) {
      if (x0[i] > 0.f) { msum += x0[i]; mcnt += 1.f; lastVal = x0[i]; lastIdx = k0 + i; }
      if (x1[i] > 0.f) { msum += x1[i]; mcnt += 1.f; lastVal = x1[i]; lastIdx = k0 + 4 + i; }
      if (x2[i] > 0.f) { msum += x2[i]; mcnt += 1.f; lastVal = x2[i]; lastIdx = k0 + 16 + i; }
      if (x3[i] > 0.f) { msum += x3[i]; mcnt += 1.f; lastVal = x3[i]; lastIdx = k0 + 20 + i; }
      a[i]      = (_Float16)x0[i];
      a[4 + i]  = (_Float16)x1[i];
      a[8 + i]  = (_Float16)x2[i];
      a[12 + i] = (_Float16)x3[i];
    }
#pragma unroll
    for (int nt = 0; nt < NT1; ++nt) {
      v16h b = *reinterpret_cast<const v16h*>(W1h + (size_t)(nt * KT1 + kt) * FRAG + lane * 16);
      c[nt] = __builtin_amdgcn_wmma_f32_16x16x32_f16(false, a, false, b, (short)0, c[nt], false, false);
    }
  }

  // lanes l and l^16 hold complementary K halves of the same row: pair-reduce
  {
    float os = __shfl_xor(msum, 16);
    float oc = __shfl_xor(mcnt, 16);
    float ov = __shfl_xor(lastVal, 16);
    int   oi = __shfl_xor(lastIdx, 16);
    msum += os; mcnt += oc;
    if (oi > lastIdx) { lastVal = ov; lastIdx = oi; }
  }

  // h = relu(x_seq @ W1 + b1) -> LDS (row-major 16x128 per wave)
#pragma unroll
  for (int nt = 0; nt < NT1; ++nt) {
    int col = nt * 16 + r;
    float bb = b1[col];
#pragma unroll
    for (int v = 0; v < 8; ++v) {
      int m = (lane < 16) ? v : v + 8;   // C-layout: VGPR v is row v / v+8
      float hv = c[nt][v] + bb;
      hs[wave][m * 128 + col] = hv > 0.f ? hv : 0.f;
    }
  }

  // layer 2: x_main = relu(h @ W2 + b2), K = 128 (A fragments re-swizzled via LDS)
  v8f c2[8] = {};
#pragma unroll
  for (int kt = 0; kt < KT2; ++kt) {
    int k0 = kt * 32 + kh * 8;
    const float* hp = &hs[wave][r * 128 + k0];
    f4 y0 = *reinterpret_cast<const f4*>(hp);
    f4 y1 = *reinterpret_cast<const f4*>(hp + 4);
    f4 y2 = *reinterpret_cast<const f4*>(hp + 16);
    f4 y3 = *reinterpret_cast<const f4*>(hp + 20);
    v16h a;
#pragma unroll
    for (int i = 0; i < 4; ++i) {
      a[i]      = (_Float16)y0[i];
      a[4 + i]  = (_Float16)y1[i];
      a[8 + i]  = (_Float16)y2[i];
      a[12 + i] = (_Float16)y3[i];
    }
#pragma unroll
    for (int nt = 0; nt < NT1; ++nt) {
      v16h b = *reinterpret_cast<const v16h*>(W2h + (size_t)(nt * KT2 + kt) * FRAG + lane * 16);
      c2[nt] = __builtin_amdgcn_wmma_f32_16x16x32_f16(false, a, false, b, (short)0, c2[nt], false, false);
    }
  }

  // x_main -> LDS (overwrite; same-wave DS is in-order), then f16 store to xcat[:,0:128]
#pragma unroll
  for (int nt = 0; nt < NT1; ++nt) {
    int col = nt * 16 + r;
    float bb = b2[col];
#pragma unroll
    for (int v = 0; v < 8; ++v) {
      int m = (lane < 16) ? v : v + 8;
      float hv = c2[nt][v] + bb;
      hs[wave][m * 128 + col] = hv > 0.f ? hv : 0.f;
    }
  }
  for (int i = lane; i < 1024; i += 32) {
    int m  = i >> 6;
    int cc = (i & 63) * 2;
    h2 val = { (_Float16)hs[wave][m * 128 + cc], (_Float16)hs[wave][m * 128 + cc + 1] };
    *reinterpret_cast<h2*>(xcat + (size_t)(Rb + m) * 192 + cc) = val;
  }

  // aux + uplink MLPs: one row per lane (lanes 0..15), write xcat[:,128:192]
  if (lane < 16) {
    float mean = msum / (mcnt + 1e-8f);
    float ltp  = (mcnt > 0.f) ? lastVal : 0.f;
    float uplink = xrow[SEQ];
    float a1[32], xa[32];
#pragma unroll
    for (int j = 0; j < 32; ++j) {
      float s = mean * w_aux1[j] + ltp * w_aux1[32 + j] + b_aux1[j];
      a1[j] = s > 0.f ? s : 0.f;
    }
#pragma unroll 4
    for (int j = 0; j < 32; ++j) {
      float s = b_aux2[j];
#pragma unroll
      for (int i = 0; i < 32; ++i) s += a1[i] * w_aux2[i * 32 + j];
      xa[j] = s > 0.f ? s : 0.f;
    }
    float u1[16], xu[16];
#pragma unroll
    for (int j = 0; j < 16; ++j) {
      float s = uplink * w_up1[j] + b_up1[j];
      u1[j] = s > 0.f ? s : 0.f;
    }
#pragma unroll 4
    for (int j = 0; j < 16; ++j) {
      float s = b_up2[j];
#pragma unroll
      for (int i = 0; i < 16; ++i) s += u1[i] * w_up2[i * 16 + j];
      xu[j] = s > 0.f ? s : 0.f;
    }
    _Float16* xr = xcat + (size_t)row * 192;
#pragma unroll
    for (int j = 0; j < 32; ++j) xr[128 + j] = (_Float16)xa[j];
#pragma unroll
    for (int j = 0; j < 16; ++j) xr[160 + j] = (_Float16)xu[j];
#pragma unroll
    for (int j = 0; j < 16; ++j) xr[176 + j] = (_Float16)0.f;   // K pad 176->192
  }
}

// ---------------------------------------------------------------------------
// Fusion head: f = relu(x_cat @ Wf1 + bf1) (K=192), out = f @ Wf2 + bf2 (K=128,N=16)
// ---------------------------------------------------------------------------
__global__ __launch_bounds__(256) void fusion_kernel(
    const _Float16* __restrict__ xcat,
    const _Float16* __restrict__ Wf1h, const float* __restrict__ bf1,
    const _Float16* __restrict__ Wf2h, const float* __restrict__ bf2,
    float* __restrict__ out)
{
  __shared__ float fs[8][16 * 128];
  const int wave = threadIdx.x >> 5;
  const int lane = threadIdx.x & 31;
  const int r    = lane & 15;
  const int kh   = lane >> 4;
  const int Rb   = (blockIdx.x * 8 + wave) * 16;
  const _Float16* xr = xcat + (size_t)(Rb + r) * 192;

  v8f c[8] = {};
#pragma unroll
  for (int kt = 0; kt < KTF1; ++kt) {
    int k0 = kt * 32 + kh * 8;
    v8h lo = *reinterpret_cast<const v8h*>(xr + k0);
    v8h hi = *reinterpret_cast<const v8h*>(xr + k0 + 16);
    v16h a;
#pragma unroll
    for (int i = 0; i < 8; ++i) { a[i] = lo[i]; a[8 + i] = hi[i]; }
#pragma unroll
    for (int nt = 0; nt < NT1; ++nt) {
      v16h b = *reinterpret_cast<const v16h*>(Wf1h + (size_t)(nt * KTF1 + kt) * FRAG + lane * 16);
      c[nt] = __builtin_amdgcn_wmma_f32_16x16x32_f16(false, a, false, b, (short)0, c[nt], false, false);
    }
  }
#pragma unroll
  for (int nt = 0; nt < NT1; ++nt) {
    int col = nt * 16 + r;
    float bb = bf1[col];
#pragma unroll
    for (int v = 0; v < 8; ++v) {
      int m = (lane < 16) ? v : v + 8;
      float hv = c[nt][v] + bb;
      fs[wave][m * 128 + col] = hv > 0.f ? hv : 0.f;
    }
  }

  v8f c2 = {};
#pragma unroll
  for (int kt = 0; kt < KTF2; ++kt) {
    int k0 = kt * 32 + kh * 8;
    const float* fp = &fs[wave][r * 128 + k0];
    f4 y0 = *reinterpret_cast<const f4*>(fp);
    f4 y1 = *reinterpret_cast<const f4*>(fp + 4);
    f4 y2 = *reinterpret_cast<const f4*>(fp + 16);
    f4 y3 = *reinterpret_cast<const f4*>(fp + 20);
    v16h a;
#pragma unroll
    for (int i = 0; i < 4; ++i) {
      a[i]      = (_Float16)y0[i];
      a[4 + i]  = (_Float16)y1[i];
      a[8 + i]  = (_Float16)y2[i];
      a[12 + i] = (_Float16)y3[i];
    }
    v16h b = *reinterpret_cast<const v16h*>(Wf2h + (size_t)kt * FRAG + lane * 16);
    c2 = __builtin_amdgcn_wmma_f32_16x16x32_f16(false, a, false, b, (short)0, c2, false, false);
  }
  float bb = bf2[r];
#pragma unroll
  for (int v = 0; v < 8; ++v) {
    int m = (lane < 16) ? v : v + 8;
    out[(size_t)(Rb + m) * 16 + r] = c2[v] + bb;
  }
}

// ---------------------------------------------------------------------------
extern "C" void kernel_launch(void* const* d_in, const int* in_sizes, int n_in,
                              void* d_out, int out_size, void* d_ws, size_t ws_size,
                              hipStream_t stream) {
  const float* x       = (const float*)d_in[0];
  const float* w_main1 = (const float*)d_in[1];
  const float* b_main1 = (const float*)d_in[2];
  const float* w_main2 = (const float*)d_in[3];
  const float* b_main2 = (const float*)d_in[4];
  const float* w_aux1  = (const float*)d_in[5];
  const float* b_aux1  = (const float*)d_in[6];
  const float* w_aux2  = (const float*)d_in[7];
  const float* b_aux2  = (const float*)d_in[8];
  const float* w_up1   = (const float*)d_in[9];
  const float* b_up1   = (const float*)d_in[10];
  const float* w_up2   = (const float*)d_in[11];
  const float* b_up2   = (const float*)d_in[12];
  const float* w_fus1  = (const float*)d_in[13];
  const float* b_fus1  = (const float*)d_in[14];
  const float* w_fus2  = (const float*)d_in[15];
  const float* b_fus2  = (const float*)d_in[16];

  int Bn = in_sizes[0] / SX;   // 65536

  // workspace layout: packed f16 weights (<600 KB) then x_cat f16 (B*192*2 = 24 MB)
  char* ws = (char*)d_ws;
  _Float16* W1h  = (_Float16*)(ws);
  _Float16* W2h  = (_Float16*)(ws + (512u << 10));
  _Float16* Wf1h = (_Float16*)(ws + (544u << 10));
  _Float16* Wf2h = (_Float16*)(ws + (592u << 10));
  _Float16* xcat = (_Float16*)(ws + (1u << 20));

  pack_kernel<<<NT1 * KT1, 32, 0, stream>>>(w_main1, W1h, SEQ, 128, KT1);
  pack_kernel<<<NT1 * KT2, 32, 0, stream>>>(w_main2, W2h, 128, 128, KT2);
  pack_kernel<<<NT1 * KTF1, 32, 0, stream>>>(w_fus1, Wf1h, 176, 128, KTF1);
  pack_kernel<<<1 * KTF2, 32, 0, stream>>>(w_fus2, Wf2h, 128, 16, KTF2);

  int grid = Bn / 128;  // 16 rows per wave * 8 waves per block
  main_kernel<<<grid, 256, 0, stream>>>(x, W1h, b_main1, W2h, b_main2,
                                        w_aux1, b_aux1, w_aux2, b_aux2,
                                        w_up1, b_up1, w_up2, b_up2, xcat);
  fusion_kernel<<<grid, 256, 0, stream>>>(xcat, Wf1h, b_fus1, Wf2h, b_fus2,
                                          (float*)d_out);
}
